// MMD_loss_24446953848866
// MI455X (gfx1250) — compile-verified, matrix operand order checked by reading, not run on hardware
//
#include <hip/hip_runtime.h>
#include <hip/hip_bf16.h>

// MMD loss on MI455X (gfx1250): split-bf16 WMMA gram + fused RBF-kernel epilogue.
// T = concat(source, target): M=8192 rows, D=512 cols, f32 row-major.
// bandwidth derived analytically: sum(L2) = 2*M*sum_sq - 2*||colsum||^2.
// Gram uses 2-term bf16 split (hi+lo): 3 WMMAs per product, ~f32 accuracy.
// hi/lo split is precomputed once into ws (if large enough) so the main
// loop is pure b128-loads + v_wmma.

#define DDIM 512
#define NROWS 4096
#define MTOT 8192
#define NT1D 64          // 8192 / 128 tiles per dimension

typedef __attribute__((ext_vector_type(16))) __bf16 v16bf;
typedef __attribute__((ext_vector_type(8)))  float  v8f;

__device__ __forceinline__ float fexp2(float x) {
#if __has_builtin(__builtin_amdgcn_exp2f)
  return __builtin_amdgcn_exp2f(x);
#else
  return exp2f(x);
#endif
}

// ---------- K1: column sums of T (for ||colsum||^2 term) ----------
__global__ void k_colsum(const float* __restrict__ src, const float* __restrict__ tgt,
                         double* __restrict__ colsumD) {
  int tid = threadIdx.x;
  int col = blockIdx.x * 64 + (tid & 63);
  int q   = tid >> 6;
  double s = 0.0;
  for (int r = q * 1024; r < (q + 1) * 1024; ++r) {
    s += (double)src[(size_t)r * DDIM + col];
    s += (double)tgt[(size_t)r * DDIM + col];
  }
  __shared__ double sh[256];
  sh[tid] = s;
  __syncthreads();
  if (tid < 64)
    colsumD[blockIdx.x * 64 + tid] = sh[tid] + sh[tid + 64] + sh[tid + 128] + sh[tid + 192];
}

// ---------- K2: per-row squared norms (one wave per row) ----------
__global__ void k_rowsq(const float* __restrict__ src, const float* __restrict__ tgt,
                        float* __restrict__ sqv) {
  int lane = threadIdx.x & 31;
  int wave = threadIdx.x >> 5;
  int row  = blockIdx.x * 8 + wave;
  const float* p = (row < NROWS) ? (src + (size_t)row * DDIM)
                                 : (tgt + (size_t)(row - NROWS) * DDIM);
  float s = 0.f;
  for (int k = lane; k < DDIM; k += 32) { float x = p[k]; s += x * x; }
#pragma unroll
  for (int off = 16; off > 0; off >>= 1) s += __shfl_xor(s, off, 32);
  if (lane == 0) sqv[row] = s;
}

// ---------- K3: bandwidth scalar -> c0 = log2(e) / (bandwidth/4) ----------
__global__ void k_stats(const double* __restrict__ colsumD,
                        const float* __restrict__ sqv, float* __restrict__ c0) {
  __shared__ double sh[256];
  int tid = threadIdx.x;
  double s = 0.0;
  for (int r = tid; r < MTOT; r += 256) s += (double)sqv[r];
  double c2 = 0.0;
  for (int d = tid; d < DDIM; d += 256) { double v = colsumD[d]; c2 += v * v; }
  sh[tid] = s; __syncthreads();
  for (int st = 128; st; st >>= 1) { if (tid < st) sh[tid] += sh[tid + st]; __syncthreads(); }
  double S = sh[0]; __syncthreads();
  sh[tid] = c2; __syncthreads();
  for (int st = 128; st; st >>= 1) { if (tid < st) sh[tid] += sh[tid + st]; __syncthreads(); }
  if (tid == 0) {
    double C = sh[0];
    double Mt = (double)MTOT;
    double sumL2 = 2.0 * Mt * S - 2.0 * C;
    double bandwidth = sumL2 / (Mt * Mt - Mt);
    double bwb = bandwidth / 4.0;                 // kernel_mul^(kernel_num//2) = 4
    c0[0] = (float)(1.4426950408889634 / bwb);    // log2(e)/bw_base
  }
}

// ---------- K_convert: one-shot f32 -> (bf16 hi, bf16 lo) split ----------
// Concatenated layout: rows 0..4095 = source, 4096..8191 = target.
__global__ void k_convert(const float* __restrict__ src, const float* __restrict__ tgt,
                          __bf16* __restrict__ hiA, __bf16* __restrict__ loA) {
  size_t idx = ((size_t)blockIdx.x * 256 + threadIdx.x) * 8;
  const size_t half = (size_t)NROWS * DDIM;
  const float* p = (idx < half) ? (src + idx) : (tgt + (idx - half));
  float4 a = *(const float4*)p;
  float4 b = *(const float4*)(p + 4);
  float f[8] = {a.x, a.y, a.z, a.w, b.x, b.y, b.z, b.w};
  union { __bf16 h[8]; float4 q; } H, L;
#pragma unroll
  for (int e = 0; e < 8; ++e) {
    __bf16 h = (__bf16)f[e];
    H.h[e] = h;
    L.h[e] = (__bf16)(f[e] - (float)h);
  }
  *(float4*)(hiA + idx) = H.q;
  *(float4*)(loA + idx) = L.q;
}

// Load one lane's 16 bf16 fragment elements: p[0..7] and p[16..23]
// (fragment element e<8 -> K=khalf+e; e>=8 -> K=16+khalf+e-8).
__device__ __forceinline__ v16bf load_frag(const __bf16* __restrict__ p) {
  union { v16bf v; float4 q[2]; } u;
  u.q[0] = *(const float4*)(p);
  u.q[1] = *(const float4*)(p + 16);
  return u.v;
}

// On-the-fly fallback: load 16 f32 and split into bf16 hi/lo.
__device__ __forceinline__ void load_split(const float* __restrict__ p,
                                           v16bf& hi, v16bf& lo) {
  float f[16];
  *(float4*)(&f[0])  = *(const float4*)(p + 0);
  *(float4*)(&f[4])  = *(const float4*)(p + 4);
  *(float4*)(&f[8])  = *(const float4*)(p + 16);
  *(float4*)(&f[12]) = *(const float4*)(p + 20);
#pragma unroll
  for (int e = 0; e < 16; ++e) {
    __bf16 h = (__bf16)f[e];
    hi[e] = h;
    lo[e] = (__bf16)(f[e] - (float)h);
  }
}

// ---------- shared epilogue: L2 -> 5-term RBF sum, block partial ----------
__device__ __forceinline__ void mmd_epilogue(
    const v8f c[2][4], const float* __restrict__ sqv, const float* __restrict__ c0p,
    double* __restrict__ partials, int pidx, int bi, int bj,
    int Rbase, int Cbase, int wRoff, int wCoff, int r16, int khalf,
    int lane, int wave) {
  const float c0 = c0p[0];
  float sqa[2][8];
#pragma unroll
  for (int fr = 0; fr < 2; ++fr) {
    const float* ps = sqv + Rbase + wRoff + fr * 16 + khalf;
    *(float4*)(&sqa[fr][0]) = *(const float4*)(ps);
    *(float4*)(&sqa[fr][4]) = *(const float4*)(ps + 4);
  }
  float sqb[4];
#pragma unroll
  for (int fc = 0; fc < 4; ++fc) sqb[fc] = sqv[Cbase + wCoff + fc * 16 + r16];

  float wgt = (bi == bj) ? 1.f : 2.f;        // symmetry: off-diagonal tiles x2
  if ((bi < 32) != (bj < 32)) wgt = -wgt;    // cross (XY/YX) blocks negative

  float lsum = 0.f;
#pragma unroll
  for (int fr = 0; fr < 2; ++fr)
#pragma unroll
    for (int fc = 0; fc < 4; ++fc)
#pragma unroll
      for (int v = 0; v < 8; ++v) {
        float g  = c[fr][fc][v];
        float l2 = sqa[fr][v] + sqb[fc] - 2.f * g;
        float na = -(l2 * c0);               // -L2*log2e/bw_base
        float ks = fexp2(na) + fexp2(0.5f * na) + fexp2(0.25f * na)
                 + fexp2(0.125f * na) + fexp2(0.0625f * na);
        lsum += ks;
      }
  lsum *= wgt;
#pragma unroll
  for (int off = 16; off > 0; off >>= 1) lsum += __shfl_xor(lsum, off, 32);

  __shared__ double wsum[8];
  if (lane == 0) wsum[wave] = (double)lsum;
  __syncthreads();
  if (threadIdx.x == 0) {
    double b = 0.0;
#pragma unroll
    for (int i = 0; i < 8; ++i) b += wsum[i];
    partials[pidx] = b;
  }
}

// ---------- K4a: main WMMA kernel, preconverted bf16 hi/lo ----------
// Grid 64x64 (bi=blockIdx.y, bj=blockIdx.x), only bj>=bi tiles live.
// 8 waves (4 row x 2 col), wave tile 32x64, block tile 128x128.
__global__ __launch_bounds__(256) void k_main_pre(
    const __bf16* __restrict__ thi, const __bf16* __restrict__ tlo,
    const float* __restrict__ sqv, const float* __restrict__ c0p,
    double* __restrict__ partials) {
  const int bj = blockIdx.x, bi = blockIdx.y;
  const int pidx = bi * NT1D + bj;
  if (bj < bi) {
    if (threadIdx.x == 0) partials[pidx] = 0.0;
    return;
  }
  const int lane  = threadIdx.x & 31;
  const int wave  = threadIdx.x >> 5;
  const int wRoff = (wave & 3) * 32;
  const int wCoff = (wave >> 2) * 64;
  const int r16   = lane & 15;
  const int khalf = (lane >> 4) << 3;
  const int Rbase = bi * 128;
  const int Cbase = bj * 128;

  const __bf16* aHi[2]; const __bf16* aLo[2];
#pragma unroll
  for (int fr = 0; fr < 2; ++fr) {
    size_t off = (size_t)(Rbase + wRoff + fr * 16 + r16) * DDIM + khalf;
    aHi[fr] = thi + off; aLo[fr] = tlo + off;
  }
  const __bf16* bHi[4]; const __bf16* bLo[4];
#pragma unroll
  for (int fc = 0; fc < 4; ++fc) {
    size_t off = (size_t)(Cbase + wCoff + fc * 16 + r16) * DDIM + khalf;
    bHi[fc] = thi + off; bLo[fc] = tlo + off;
  }

  v8f c[2][4] = {};
  for (int k0 = 0; k0 < DDIM; k0 += 32) {
    v16bf ah[2], al[2], bh[4], bl[4];
#pragma unroll
    for (int fr = 0; fr < 2; ++fr) { ah[fr] = load_frag(aHi[fr] + k0); al[fr] = load_frag(aLo[fr] + k0); }
#pragma unroll
    for (int fc = 0; fc < 4; ++fc) { bh[fc] = load_frag(bHi[fc] + k0); bl[fc] = load_frag(bLo[fc] + k0); }
#pragma unroll
    for (int fr = 0; fr < 2; ++fr)
#pragma unroll
      for (int fc = 0; fc < 4; ++fc) {
        c[fr][fc] = __builtin_amdgcn_wmma_f32_16x16x32_bf16(
            false, ah[fr], false, bh[fc], (short)0, c[fr][fc], false, false);
        c[fr][fc] = __builtin_amdgcn_wmma_f32_16x16x32_bf16(
            false, ah[fr], false, bl[fc], (short)0, c[fr][fc], false, false);
        c[fr][fc] = __builtin_amdgcn_wmma_f32_16x16x32_bf16(
            false, al[fr], false, bh[fc], (short)0, c[fr][fc], false, false);
      }
  }
  mmd_epilogue(c, sqv, c0p, partials, pidx, bi, bj, Rbase, Cbase,
               wRoff, wCoff, r16, khalf, lane, wave);
}

// ---------- K4b: fallback, on-the-fly split (small ws) ----------
__global__ __launch_bounds__(256) void k_main_fly(
    const float* __restrict__ src, const float* __restrict__ tgt,
    const float* __restrict__ sqv, const float* __restrict__ c0p,
    double* __restrict__ partials) {
  const int bj = blockIdx.x, bi = blockIdx.y;
  const int pidx = bi * NT1D + bj;
  if (bj < bi) {
    if (threadIdx.x == 0) partials[pidx] = 0.0;
    return;
  }
  const int lane  = threadIdx.x & 31;
  const int wave  = threadIdx.x >> 5;
  const int wRoff = (wave & 3) * 32;
  const int wCoff = (wave >> 2) * 64;
  const int r16   = lane & 15;
  const int khalf = (lane >> 4) << 3;
  const int Rbase = bi * 128;
  const int Cbase = bj * 128;
  const float* Abase = (bi < 32) ? (src + (size_t)Rbase * DDIM)
                                 : (tgt + (size_t)(Rbase - NROWS) * DDIM);
  const float* Bbase = (bj < 32) ? (src + (size_t)Cbase * DDIM)
                                 : (tgt + (size_t)(Cbase - NROWS) * DDIM);
  const float* aRow[2];
  aRow[0] = Abase + (size_t)(wRoff + r16) * DDIM;
  aRow[1] = aRow[0] + (size_t)16 * DDIM;
  const float* bRow[4];
#pragma unroll
  for (int fc = 0; fc < 4; ++fc)
    bRow[fc] = Bbase + (size_t)(wCoff + fc * 16 + r16) * DDIM;

  v8f c[2][4] = {};
  for (int k0 = 0; k0 < DDIM; k0 += 32) {
    v16bf ah[2], al[2], bh[4], bl[4];
#pragma unroll
    for (int fr = 0; fr < 2; ++fr) load_split(aRow[fr] + k0 + khalf, ah[fr], al[fr]);
#pragma unroll
    for (int fc = 0; fc < 4; ++fc) load_split(bRow[fc] + k0 + khalf, bh[fc], bl[fc]);
#pragma unroll
    for (int fr = 0; fr < 2; ++fr)
#pragma unroll
      for (int fc = 0; fc < 4; ++fc) {
        c[fr][fc] = __builtin_amdgcn_wmma_f32_16x16x32_bf16(
            false, ah[fr], false, bh[fc], (short)0, c[fr][fc], false, false);
        c[fr][fc] = __builtin_amdgcn_wmma_f32_16x16x32_bf16(
            false, ah[fr], false, bl[fc], (short)0, c[fr][fc], false, false);
        c[fr][fc] = __builtin_amdgcn_wmma_f32_16x16x32_bf16(
            false, al[fr], false, bh[fc], (short)0, c[fr][fc], false, false);
      }
  }
  mmd_epilogue(c, sqv, c0p, partials, pidx, bi, bj, Rbase, Cbase,
               wRoff, wCoff, r16, khalf, lane, wave);
}

// ---------- K5: deterministic final reduction ----------
__global__ void k_final(const double* __restrict__ partials, float* __restrict__ out) {
  __shared__ double sh[256];
  int tid = threadIdx.x;
  double s = 0.0;
  for (int i = tid; i < NT1D * NT1D; i += 256) s += partials[i];
  sh[tid] = s; __syncthreads();
  for (int st = 128; st; st >>= 1) { if (tid < st) sh[tid] += sh[tid + st]; __syncthreads(); }
  if (tid == 0) out[0] = (float)(sh[0] / ((double)NROWS * (double)NROWS));
}

extern "C" void kernel_launch(void* const* d_in, const int* in_sizes, int n_in,
                              void* d_out, int out_size, void* d_ws, size_t ws_size,
                              hipStream_t stream) {
  (void)in_sizes; (void)n_in; (void)out_size;
  const float* src = (const float*)d_in[0];
  const float* tgt = (const float*)d_in[1];
  char* w = (char*)d_ws;
  float*  c0       = (float*)(w + 0);          // 4 B (pad to 64)
  double* colsumD  = (double*)(w + 64);        // 512*8  = 4096 B
  float*  sqv      = (float*)(w + 4160);       // 8192*4 = 32768 B
  double* partials = (double*)(w + 36928);     // 4096*8 = 32768 B (ends 69696)
  __bf16* thi      = (__bf16*)(w + 131072);    // 8192*512*2 = 8 MiB
  __bf16* tlo      = (__bf16*)(w + 8519680);   // 8 MiB (ends 16908288)
  const size_t PRE_NEED = 16908288;
  float* out = (float*)d_out;

  hipLaunchKernelGGL(k_colsum, dim3(8),    dim3(256), 0, stream, src, tgt, colsumD);
  hipLaunchKernelGGL(k_rowsq,  dim3(1024), dim3(256), 0, stream, src, tgt, sqv);
  hipLaunchKernelGGL(k_stats,  dim3(1),    dim3(256), 0, stream, colsumD, sqv, c0);
  if (ws_size >= PRE_NEED) {
    hipLaunchKernelGGL(k_convert, dim3(2048), dim3(256), 0, stream, src, tgt, thi, tlo);
    hipLaunchKernelGGL(k_main_pre, dim3(NT1D, NT1D), dim3(256), 0, stream,
                       thi, tlo, sqv, c0, partials);
  } else {
    hipLaunchKernelGGL(k_main_fly, dim3(NT1D, NT1D), dim3(256), 0, stream,
                       src, tgt, sqv, c0, partials);
  }
  hipLaunchKernelGGL(k_final, dim3(1), dim3(256), 0, stream, partials, out);
}
